// LlavaViTAttention_23252952940582
// MI455X (gfx1250) — matrix operand order repaired and checked by reading, not compiled, not run on hardware
//
#include <hip/hip_runtime.h>

#define B_  16
#define S_  784
#define D_  768
#define H_  12
#define DH_ 64

typedef __attribute__((ext_vector_type(16))) __bf16 v16bf;
typedef __attribute__((ext_vector_type(8)))  __bf16 v8bf;
typedef __attribute__((ext_vector_type(8)))  float  v8f;

// float -> bf16 bits, round-to-nearest-even (epilogues / one-time converts only)
__device__ __forceinline__ unsigned bf_bits(float f) {
    unsigned u = __builtin_bit_cast(unsigned, f);
    u += 0x7FFFu + ((u >> 16) & 1u);
    return u >> 16;
}
// 8 contiguous bf16 -> v8bf (single b128 load, no repack)
__device__ __forceinline__ v8bf ld8(const unsigned short* p) {
    return __builtin_bit_cast(v8bf, *(const uint4*)p);
}
__device__ __forceinline__ v16bf frag(v8bf lo, v8bf hi) {
    return __builtin_shufflevector(lo, hi, 0,1,2,3,4,5,6,7,8,9,10,11,12,13,14,15);
}
__device__ __forceinline__ v8f wmma_bf16(v16bf a, v16bf b, v8f c) {
    return __builtin_amdgcn_wmma_f32_16x16x32_bf16(false, a, false, b, (short)0, c, false, false);
}

// ---------------------------------------------------------------------------
// Kernel 0a: fp32 -> bf16 elementwise (X, weights), once.
// ---------------------------------------------------------------------------
__global__ __launch_bounds__(256)
void cvt_kernel(const float* __restrict__ src, unsigned short* __restrict__ dst, int n4) {
    int i = blockIdx.x * blockDim.x + threadIdx.x;
    if (i >= n4) return;
    float4 f = ((const float4*)src)[i];
    uint2 o;
    o.x = bf_bits(f.x) | (bf_bits(f.y) << 16);
    o.y = bf_bits(f.z) | (bf_bits(f.w) << 16);
    ((uint2*)dst)[i] = o;
}

// ---------------------------------------------------------------------------
// Kernel 0b: rope angles -> cos/sin fp32 tables, once.
// ---------------------------------------------------------------------------
__global__ __launch_bounds__(256)
void rope_tab_kernel(const float* __restrict__ rope,
                     float* __restrict__ cosT, float* __restrict__ sinT, int n) {
    int i = blockIdx.x * blockDim.x + threadIdx.x;
    if (i >= n) return;
    float sn, cs;
    __sincosf(rope[i], &sn, &cs);
    cosT[i] = cs;
    sinT[i] = sn;
}

// ---------------------------------------------------------------------------
// Kernel 1: Q/K/V = X @ W^T + b (+RoPE on Q,K). One wave = 32x64 tile (1 head).
// grid = (392 row tiles, 12 heads, 3 matrices), block = 32.
// Q,K -> bf16 [B,H,S,DH]; V -> bf16 transposed [B,H,DH,S].
// ---------------------------------------------------------------------------
__global__ __launch_bounds__(32)
void qkv_rope_kernel(const unsigned short* __restrict__ xbf,
                     const float* __restrict__ cosT, const float* __restrict__ sinT,
                     const unsigned short* __restrict__ wqb, const float* __restrict__ bq,
                     const unsigned short* __restrict__ wkb, const float* __restrict__ bk,
                     const unsigned short* __restrict__ wvb, const float* __restrict__ bv,
                     unsigned short* __restrict__ qbuf,
                     unsigned short* __restrict__ kbuf,
                     unsigned short* __restrict__ vtbuf) {
    const int lane = threadIdx.x;
    const int n    = lane & 15;
    const int half = lane >> 4;
    const int mt   = blockIdx.x;     // 392 tiles of 32 rows over B*S
    const int nt   = blockIdx.y;     // head (64-wide N tile)
    const int which = blockIdx.z;    // 0=q 1=k 2=v

    const unsigned short* w; const float* bias;
    if (which == 0)      { w = wqb; bias = bq; }
    else if (which == 1) { w = wkb; bias = bk; }
    else                 { w = wvb; bias = bv; }

    const int g0 = mt * 32;                                       // global row base
    const unsigned short* xrow0 = xbf + (size_t)(g0 + n) * D_;    // A rows (tile 0)
    const unsigned short* xrow1 = xbf + (size_t)(g0 + 16 + n) * D_;
    const unsigned short* wrows[4];
#pragma unroll
    for (int t = 0; t < 4; ++t)
        wrows[t] = w + (size_t)(nt * 64 + t * 16 + n) * D_;       // B col = e

    const v8f vz = {0.f,0.f,0.f,0.f,0.f,0.f,0.f,0.f};
    v8f acc[8] = {vz, vz, vz, vz, vz, vz, vz, vz};                // [mh*4 + t]

#pragma unroll 2
    for (int kk = 0; kk < D_; kk += 32) {
        v16bf a0 = frag(ld8(xrow0 + kk + 8 * half), ld8(xrow0 + kk + 16 + 8 * half));
        v16bf a1 = frag(ld8(xrow1 + kk + 8 * half), ld8(xrow1 + kk + 16 + 8 * half));
#pragma unroll
        for (int t = 0; t < 4; ++t) {
            const unsigned short* wr = wrows[t] + kk + 8 * half;
            v16bf bfv = frag(ld8(wr), ld8(wr + 16));
            acc[t]     = wmma_bf16(a0, bfv, acc[t]);
            acc[4 + t] = wmma_bf16(a1, bfv, acc[4 + t]);
        }
    }

    float be[4];
#pragma unroll
    for (int t = 0; t < 4; ++t) be[t] = bias[nt * 64 + t * 16 + n];

    if (which < 2) {
        unsigned short* ob = (which == 0) ? qbuf : kbuf;
#pragma unroll
        for (int mh = 0; mh < 2; ++mh) {
            const int gb = g0 + mh * 16 + 8 * half;   // base row of this 8-row group
            const int b  = gb / S_;
            const int sb = gb - b * S_;               // group never crosses a batch
            const size_t qk0 = ((size_t)(b * H_ + nt) * S_ + sb) * DH_;
            const size_t rp0 = ((size_t)b * S_ + sb) * DH_;
#pragma unroll
            for (int t = 0; t < 4; ++t) {
                const int dh = t * 16 + n;
#pragma unroll
                for (int r = 0; r < 8; ++r) {
                    float y = acc[mh * 4 + t][r] + be[t];
                    float part = __shfl_xor(y, 1, 32);            // rope pair partner
                    float rh = (n & 1) ? part : -part;            // rotate_half
                    const size_t ri = rp0 + (size_t)r * DH_ + dh;
                    float outv = y * cosT[ri] + rh * sinT[ri];
                    ob[qk0 + (size_t)r * DH_ + dh] = (unsigned short)bf_bits(outv);
                }
            }
        }
    } else {
#pragma unroll
        for (int mh = 0; mh < 2; ++mh) {
            const int gb = g0 + mh * 16 + 8 * half;
            const int b  = gb / S_;
            const int sb = gb - b * S_;
#pragma unroll
            for (int t = 0; t < 4; ++t) {
                const int dh = t * 16 + n;
                unsigned v[8];
#pragma unroll
                for (int r = 0; r < 8; ++r) v[r] = bf_bits(acc[mh * 4 + t][r] + be[t]);
                uint4 pk;
                pk.x = v[0] | (v[1] << 16);
                pk.y = v[2] | (v[3] << 16);
                pk.z = v[4] | (v[5] << 16);
                pk.w = v[6] | (v[7] << 16);
                size_t base = ((size_t)(b * H_ + nt) * DH_ + dh) * S_ + sb;
                *(uint4*)(vtbuf + base) = pk;
            }
        }
    }
}

// ---------------------------------------------------------------------------
// Kernel 2: flash attention, 64-key chunks. One wave = 16 query rows, one (b,h).
// grid = (49 q-tiles, 12 heads, 16 batch), block = 32.
// ---------------------------------------------------------------------------
__global__ __launch_bounds__(32)
void attn_kernel(const unsigned short* __restrict__ qbuf,
                 const unsigned short* __restrict__ kbuf,
                 const unsigned short* __restrict__ vtbuf,
                 unsigned short* __restrict__ obuf) {
    __shared__ __align__(16) unsigned short plds[16 * 64];   // 2KB P staging

    const int lane = threadIdx.x;
    const int n    = lane & 15;
    const int half = lane >> 4;
    const int qt   = blockIdx.x;
    const int bh   = blockIdx.z * H_ + blockIdx.y;

    const unsigned short* Q  = qbuf  + (size_t)bh * S_ * DH_;
    const unsigned short* K  = kbuf  + (size_t)bh * S_ * DH_;
    const unsigned short* VT = vtbuf + (size_t)bh * DH_ * S_;

    const unsigned short* qrow = Q + (size_t)(qt * 16 + n) * DH_;
    const v16bf qa0 = frag(ld8(qrow + 16 * half),      ld8(qrow + 8 + 16 * half));
    const v16bf qa1 = frag(ld8(qrow + 32 + 16 * half), ld8(qrow + 40 + 16 * half));

    float rm[8], rl[8];
#pragma unroll
    for (int r = 0; r < 8; ++r) { rm[r] = -1.0e30f; rl[r] = 0.f; }
    const v8f vz = {0.f,0.f,0.f,0.f,0.f,0.f,0.f,0.f};
    v8f o[4] = {vz, vz, vz, vz};
    const float scale = 0.125f;   // DH^-0.5

    for (int cb = 0; cb < 832; cb += 64) {        // 13 chunks of 64 keys
        // ---- scores: four 16-key tiles ----
        v8f sc[4] = {vz, vz, vz, vz};
        bool vmask[4];
#pragma unroll
        for (int t = 0; t < 4; ++t) {
            const int key = cb + t * 16 + n;
            vmask[t] = key < S_;
            const unsigned short* kr = K + (size_t)min(key, S_ - 1) * DH_ + 16 * half;
            sc[t] = wmma_bf16(qa0, frag(ld8(kr),      ld8(kr + 8)),  sc[t]);
            sc[t] = wmma_bf16(qa1, frag(ld8(kr + 32), ld8(kr + 40)), sc[t]);
        }

        // ---- online softmax (rows replicated across each 16-lane half) ----
#pragma unroll
        for (int r = 0; r < 8; ++r) {
            float a[4];
#pragma unroll
            for (int t = 0; t < 4; ++t)
                a[t] = vmask[t] ? sc[t][r] * scale : -1.0e30f;
            float mx = fmaxf(fmaxf(a[0], a[1]), fmaxf(a[2], a[3]));
            mx = fmaxf(mx, __shfl_xor(mx, 1, 32));
            mx = fmaxf(mx, __shfl_xor(mx, 2, 32));
            mx = fmaxf(mx, __shfl_xor(mx, 4, 32));
            mx = fmaxf(mx, __shfl_xor(mx, 8, 32));
            float mnew  = fmaxf(rm[r], mx);
            float alpha = __expf(rm[r] - mnew);
            rm[r] = mnew;
            float p[4];
#pragma unroll
            for (int t = 0; t < 4; ++t) p[t] = __expf(a[t] - mnew);
            float rs = (p[0] + p[1]) + (p[2] + p[3]);
            rs += __shfl_xor(rs, 1, 32);
            rs += __shfl_xor(rs, 2, 32);
            rs += __shfl_xor(rs, 4, 32);
            rs += __shfl_xor(rs, 8, 32);
            rl[r] = rl[r] * alpha + rs;
            o[0][r] *= alpha; o[1][r] *= alpha; o[2][r] *= alpha; o[3][r] *= alpha;
            const int m = r + 8 * half;
#pragma unroll
            for (int t = 0; t < 4; ++t)
                plds[m * 64 + t * 16 + n] = (unsigned short)bf_bits(p[t]);
        }

        // ---- O += P @ V : two 32-key k-steps, 4 dh-tiles each ----
#pragma unroll
        for (int ks = 0; ks < 2; ++ks) {
            const v16bf pa = frag(ld8(&plds[n * 64 + ks * 32 + 16 * half]),
                                  ld8(&plds[n * 64 + ks * 32 + 8 + 16 * half]));
            const int g = min(cb + ks * 32 + 16 * half, S_ - 16);
#pragma unroll
            for (int t = 0; t < 4; ++t) {
                const unsigned short* vrow = VT + (size_t)(t * 16 + n) * S_ + g;
                o[t] = wmma_bf16(pa, frag(ld8(vrow), ld8(vrow + 8)), o[t]);
            }
        }
    }

#pragma unroll
    for (int r = 0; r < 8; ++r) {
        const int s = qt * 16 + r + 8 * half;
        const float inv = 1.0f / rl[r];
        unsigned short* orow = obuf + ((size_t)bh * S_ + s) * DH_;
#pragma unroll
        for (int t = 0; t < 4; ++t)
            orow[t * 16 + n] = (unsigned short)bf_bits(o[t][r] * inv);
    }
}

// ---------------------------------------------------------------------------
// Kernel 3: out = ctx @ Wo^T + bo. One wave = 32x64 tile.
// grid = (392 row tiles, 12 col tiles), block = 32.
// ---------------------------------------------------------------------------
__global__ __launch_bounds__(32)
void proj_kernel(const unsigned short* __restrict__ ctx,   // obuf [B,H,S,DH]
                 const unsigned short* __restrict__ wob, const float* __restrict__ bo,
                 float* __restrict__ out) {
    const int lane = threadIdx.x;
    const int n    = lane & 15;
    const int half = lane >> 4;
    const int mt   = blockIdx.x;            // 392 tiles of 32 rows
    const int nt   = blockIdx.y;            // 12 tiles of 64 output features

    const int g0 = mt * 32;
    // per-lane A-row bases in ctx ([B,H,S,DH] viewed as [B,S,D])
    const int gr0 = g0 + n,       b0 = gr0 / S_, s0r = gr0 - b0 * S_;
    const int gr1 = g0 + 16 + n,  b1 = gr1 / S_, s1r = gr1 - b1 * S_;
    const size_t cbase0 = ((size_t)b0 * H_) * S_ * DH_ + (size_t)s0r * DH_;
    const size_t cbase1 = ((size_t)b1 * H_) * S_ * DH_ + (size_t)s1r * DH_;

    const unsigned short* wrows[4];
#pragma unroll
    for (int t = 0; t < 4; ++t)
        wrows[t] = wob + (size_t)(nt * 64 + t * 16 + n) * D_;

    const v8f vz = {0.f,0.f,0.f,0.f,0.f,0.f,0.f,0.f};
    v8f acc[8] = {vz, vz, vz, vz, vz, vz, vz, vz};

#pragma unroll 2
    for (int kk = 0; kk < D_; kk += 32) {
        const int k0 = kk + 8 * half;       // 8-elem groups stay inside one head
        const int k1 = kk + 16 + 8 * half;
        const size_t o0 = (size_t)(k0 >> 6) * S_ * DH_ + (k0 & 63);
        const size_t o1 = (size_t)(k1 >> 6) * S_ * DH_ + (k1 & 63);
        v16bf a0 = frag(ld8(ctx + cbase0 + o0), ld8(ctx + cbase0 + o1));
        v16bf a1 = frag(ld8(ctx + cbase1 + o0), ld8(ctx + cbase1 + o1));
#pragma unroll
        for (int t = 0; t < 4; ++t) {
            const unsigned short* wr = wrows[t] + kk + 8 * half;
            v16bf bfv = frag(ld8(wr), ld8(wr + 16));
            acc[t]     = wmma_bf16(a0, bfv, acc[t]);
            acc[4 + t] = wmma_bf16(a1, bfv, acc[4 + t]);
        }
    }

#pragma unroll
    for (int mh = 0; mh < 2; ++mh) {
        const int gb = g0 + mh * 16 + 8 * half;
#pragma unroll
        for (int t = 0; t < 4; ++t) {
            const int e  = nt * 64 + t * 16 + n;
            const float be = bo[e];
#pragma unroll
            for (int r = 0; r < 8; ++r)
                out[(size_t)(gb + r) * D_ + e] = acc[mh * 4 + t][r] + be;
        }
    }
}

// ---------------------------------------------------------------------------
extern "C" void kernel_launch(void* const* d_in, const int* in_sizes, int n_in,
                              void* d_out, int out_size, void* d_ws, size_t ws_size,
                              hipStream_t stream) {
    (void)in_sizes; (void)n_in; (void)out_size; (void)ws_size;
    const float* x    = (const float*)d_in[0];
    const float* rope = (const float*)d_in[1];
    const float* wq   = (const float*)d_in[2];
    const float* bq   = (const float*)d_in[3];
    const float* wk   = (const float*)d_in[4];
    const float* bk   = (const float*)d_in[5];
    const float* wv   = (const float*)d_in[6];
    const float* bv   = (const float*)d_in[7];
    const float* wo   = (const float*)d_in[8];
    const float* bo   = (const float*)d_in[9];
    float* out = (float*)d_out;

    const size_t act  = (size_t)B_ * H_ * S_ * DH_;  // 9,633,792 (= B*S*D)
    const size_t wsz  = (size_t)D_ * D_;             // 589,824
    const size_t rpsz = (size_t)B_ * S_ * DH_;       // 802,816
    unsigned short* qbuf  = (unsigned short*)d_ws;
    unsigned short* kbuf  = qbuf  + act;
    unsigned short* vtbuf = kbuf  + act;
    unsigned short* obuf  = vtbuf + act;
    unsigned short* xbf   = obuf  + act;
    unsigned short* wqb   = xbf   + act;
    unsigned short* wkb   = wqb   + wsz;
    unsigned short* wvb   = wkb   + wsz;
    unsigned short* wob   = wvb   + wsz;
    float* cosT = (float*)(wob + wsz);
    float* sinT = cosT + rpsz;

    const int xn4 = (int)(act / 4), wn4 = (int)(wsz / 4);
    cvt_kernel<<<(xn4 + 255) / 256, 256, 0, stream>>>(x,  xbf, xn4);
    cvt_kernel<<<(wn4 + 255) / 256, 256, 0, stream>>>(wq, wqb, wn4);
    cvt_kernel<<<(wn4 + 255) / 256, 256, 0, stream>>>(wk, wkb, wn4);
    cvt_kernel<<<(wn4 + 255) / 256, 256, 0, stream>>>(wv, wvb, wn4);
    cvt_kernel<<<(wn4 + 255) / 256, 256, 0, stream>>>(wo, wob, wn4);
    rope_tab_kernel<<<((int)rpsz + 255) / 256, 256, 0, stream>>>(rope, cosT, sinT, (int)rpsz);

    qkv_rope_kernel<<<dim3((B_ * S_) / 32, H_, 3), 32, 0, stream>>>(
        xbf, cosT, sinT, wqb, bq, wkb, bk, wvb, bv, qbuf, kbuf, vtbuf);
    attn_kernel<<<dim3(S_ / 16, H_, B_), 32, 0, stream>>>(qbuf, kbuf, vtbuf, obuf);
    proj_kernel<<<dim3((B_ * S_) / 32, H_), 32, 0, stream>>>(obuf, wob, bo, out);
}
